// EngineeringGNN_45028437131700
// MI455X (gfx1250) — compile-verified, matrix operand order checked by reading, not run on hardware
//
#include <hip/hip_runtime.h>
#include <hip/hip_bf16.h>
#include <math.h>

// ---------------------------------------------------------------------------
// EngineeringGNN forward for gfx1250 (MI455X, wave32, WMMA).
//
// K=128 GEMMs on v_wmma_f32_16x16x32_f16 (f16 in, f32 accum). Weights are
// pre-swizzled into WMMA B-fragment order in LDS (fragment load = 2x
// ds_load_b128). Each wave owns a 16-row block, keeps the 4 A K-fragments in
// registers, and sweeps all N-tiles (A read 1x from global).
// All GEMM calls here have M % 16 == 0 (NN=100000, NE=500000), so the tile
// epilogue is branch-free.
// Edge streams stay fp32 (bandwidth bottleneck ~5-6 GB -> ~250us @23.3TB/s).
// ---------------------------------------------------------------------------

#define NN 100000   // nodes
#define NE 500000   // edges
#define HD 128

typedef __attribute__((ext_vector_type(16))) _Float16 v16h;
typedef __attribute__((ext_vector_type(8)))  float    v8f;

__device__ __forceinline__ float warp_sum32(float v) {
#pragma unroll
  for (int m = 16; m >= 1; m >>= 1) v += __shfl_xor(v, m, 32);
  return v;
}

__device__ __forceinline__ float softplus_f(float x) {
  return x > 20.f ? x : log1pf(expf(x));
}

// ---------------------------------------------------------------------------
// Generic GEMM: C[M,Ncols] = act(A[M,128] @ W[128,Ncols] + bias), K fixed 128.
// REQUIRES: M % 16 == 0.
//
// LDS layout of W (pre-swizzled to B-fragment order):
//   Wl[ ((tn*4 + kk)*32 + lane)*16 + j ] = (f16) W[k][n]
//   with k = kk*32 + 8*(lane>>4) + (j&7) + 16*(j>>3),  n = tn*16 + (lane&15)
// so a lane's fragment is 16 contiguous halves (32B) -> 2x ds_load_b128.
//
// A fragment (16x32 f16, ISA 7.12.2): lane%16 = M row,
//   element j -> k = 8*(lane/16) + (j&7) + 16*(j>>3).
// C/D (16x16 f32): element v -> row v + 8*(lane/16), col lane%16.
// ---------------------------------------------------------------------------
__global__ __launch_bounds__(256)
void gemm_k128_wmma(const float* __restrict__ A, const float* __restrict__ W,
                    const float* __restrict__ bias, float* __restrict__ C,
                    int M, int Ncols, int relu) {
  __shared__ _Float16 Wl[128 * 128];
  const int tid = threadIdx.x;
  const int total = 128 * Ncols;
  for (int i = tid; i < total; i += 256) {
    const int j  = i & 15;
    const int ln = (i >> 4) & 31;
    const int kk = (i >> 9) & 3;
    const int tn = i >> 11;
    const int k  = kk * 32 + ((ln >> 4) << 3) + (j & 7) + ((j >> 3) << 4);
    const int n  = (tn << 4) + (ln & 15);
    Wl[i] = (_Float16)W[k * Ncols + n];
  }
  __syncthreads();

  const int lane = tid & 31;
  const int hi   = lane >> 4;
  const int lo   = lane & 15;
  const int tilesM = M >> 4;            // exact: M % 16 == 0
  const int tilesN = Ncols >> 4;

  for (int tm = blockIdx.x * 8 + (tid >> 5); tm < tilesM; tm += gridDim.x * 8) {
    const int row = tm * 16 + lo;
    const float* Ar = A + (size_t)row * 128;
    v16h afrag[4];
#pragma unroll
    for (int kk = 0; kk < 4; ++kk) {
      const float* Ap = Ar + kk * 32 + hi * 8;
#pragma unroll
      for (int j = 0; j < 8; ++j) {
        afrag[kk][j]     = (_Float16)Ap[j];
        afrag[kk][j + 8] = (_Float16)Ap[j + 16];
      }
    }

    for (int tn = 0; tn < tilesN; ++tn) {
      v8f acc = {};
#pragma unroll
      for (int kk = 0; kk < 4; ++kk) {
        const v16h b = *(const v16h*)(Wl + (((tn * 4 + kk) * 32 + lane) << 4));
        acc = __builtin_amdgcn_wmma_f32_16x16x32_f16(false, afrag[kk], false, b,
                                                     (short)0, acc, false, false);
      }
      const int col = tn * 16 + lo;
      const float bc = bias[col];
      float* Cp = C + (size_t)(tm * 16 + 8 * hi) * Ncols + col;
#pragma unroll
      for (int v = 0; v < 8; ++v) {
        float y = acc[v] + bc;
        if (relu) y = fmaxf(y, 0.f);
        Cp[(size_t)v * Ncols] = y;
      }
    }
  }
}

// ---------------------------------------------------------------------------
// LayerNorm over 128 cols, one wave per row. Optional: add pf into first 32
// cols AFTER LN (node_enc path), optional ReLU AFTER LN (conv path).
// ---------------------------------------------------------------------------
__global__ __launch_bounds__(256)
void ln_kernel(const float* __restrict__ X, float* __restrict__ Y,
               const float* __restrict__ g, const float* __restrict__ be,
               const float* __restrict__ pf, int M, int relu) {
  const int row  = (int)((blockIdx.x * 256 + threadIdx.x) >> 5);
  const int lane = threadIdx.x & 31;
  if (row >= M) return;
  const int c = lane * 4;
  const float4 xv = *(const float4*)(X + (size_t)row * 128 + c);
  float s = warp_sum32(xv.x + xv.y + xv.z + xv.w);
  const float mu = s * (1.f / 128.f);
  const float dx = xv.x - mu, dy = xv.y - mu, dz = xv.z - mu, dw = xv.w - mu;
  float var = warp_sum32(dx * dx + dy * dy + dz * dz + dw * dw) * (1.f / 128.f);
  const float rs = rsqrtf(var + 1e-5f);
  float4 o;
  o.x = dx * rs * g[c + 0] + be[c + 0];
  o.y = dy * rs * g[c + 1] + be[c + 1];
  o.z = dz * rs * g[c + 2] + be[c + 2];
  o.w = dw * rs * g[c + 3] + be[c + 3];
  if (pf != nullptr && c < 32) {
    const float4 p = *(const float4*)(pf + (size_t)row * 32 + c);
    o.x += p.x; o.y += p.y; o.z += p.z; o.w += p.w;
  }
  if (relu) {
    o.x = fmaxf(o.x, 0.f); o.y = fmaxf(o.y, 0.f);
    o.z = fmaxf(o.z, 0.f); o.w = fmaxf(o.w, 0.f);
  }
  *(float4*)(Y + (size_t)row * 128 + c) = o;
}

// ---------------------------------------------------------------------------
// Small-K linear + ReLU (din = 5 or 6, Nout = 128): one thread per output.
// ---------------------------------------------------------------------------
__global__ __launch_bounds__(256)
void small_linear_relu(const float* __restrict__ X, const float* __restrict__ W,
                       const float* __restrict__ b, float* __restrict__ Y,
                       int M, int din) {
  const long long i = (long long)blockIdx.x * 256 + threadIdx.x;
  if (i >= (long long)M * 128) return;
  const int r = (int)(i >> 7), c = (int)(i & 127);
  float acc = b[c];
  const float* x = X + (size_t)r * din;
  for (int k = 0; k < din; ++k) acc = fmaf(x[k], W[k * 128 + c], acc);
  Y[i] = fmaxf(acc, 0.f);
}

// ---------------------------------------------------------------------------
// phys head: pf[N,32] = relu(pos @ W1[3,32] + b1) @ W2[32,32] + b2.
// One wave per node; cross-lane via __shfl broadcast.
// ---------------------------------------------------------------------------
__global__ __launch_bounds__(256)
void phys_kernel(const float* __restrict__ pos,
                 const float* __restrict__ W1, const float* __restrict__ b1,
                 const float* __restrict__ W2, const float* __restrict__ b2,
                 float* __restrict__ pf, int M) {
  const int row  = (int)((blockIdx.x * 256 + threadIdx.x) >> 5);
  const int lane = threadIdx.x & 31;
  if (row >= M) return;
  const float p0 = pos[(size_t)row * 3 + 0];
  const float p1 = pos[(size_t)row * 3 + 1];
  const float p2 = pos[(size_t)row * 3 + 2];
  float h1 = b1[lane];
  h1 = fmaf(p0, W1[lane], h1);
  h1 = fmaf(p1, W1[32 + lane], h1);
  h1 = fmaf(p2, W1[64 + lane], h1);
  h1 = fmaxf(h1, 0.f);
  float acc = b2[lane];
#pragma unroll
  for (int k = 0; k < 32; ++k) {
    const float hk = __shfl(h1, k, 32);
    acc = fmaf(hk, W2[k * 32 + lane], acc);
  }
  pf[(size_t)row * 32 + lane] = acc;
}

// ---------------------------------------------------------------------------
// GINE message + aggregate: aggr[dst] += relu(h[src] + ea[e]).
// One wave per edge, float4 per lane, 4 fp32 global atomics per lane.
// ---------------------------------------------------------------------------
__global__ __launch_bounds__(256)
void message_kernel(const float* __restrict__ h, const float* __restrict__ ea,
                    const int* __restrict__ src, const int* __restrict__ dst,
                    float* __restrict__ aggr, int E) {
  const int e    = (int)((blockIdx.x * 256 + threadIdx.x) >> 5);
  const int lane = threadIdx.x & 31;
  if (e >= E) return;
  const int s = src[e], d = dst[e];
  const int c = lane * 4;
  const float4 hv = *(const float4*)(h + (size_t)s * 128 + c);
  const float4 ev = *(const float4*)(ea + (size_t)e * 128 + c);
  const float m0 = fmaxf(hv.x + ev.x, 0.f);
  const float m1 = fmaxf(hv.y + ev.y, 0.f);
  const float m2 = fmaxf(hv.z + ev.z, 0.f);
  const float m3 = fmaxf(hv.w + ev.w, 0.f);
  float* out = aggr + (size_t)d * 128 + c;
  atomicAdd(out + 0, m0);
  atomicAdd(out + 1, m1);
  atomicAdd(out + 2, m2);
  atomicAdd(out + 3, m3);
}

// ---------------------------------------------------------------------------
// Head second layer: Y[M,dout] = (softplus?)(Hin[M,64] @ W[64,dout] + b).
// ---------------------------------------------------------------------------
__global__ __launch_bounds__(256)
void head_l2_kernel(const float* __restrict__ Hin, const float* __restrict__ W,
                    const float* __restrict__ b, float* __restrict__ Y,
                    int M, int dout, int use_softplus) {
  const int r = blockIdx.x * 256 + threadIdx.x;
  if (r >= M) return;
  const float* x = Hin + (size_t)r * 64;
  for (int o = 0; o < dout; ++o) {
    float acc = b[o];
    for (int k = 0; k < 64; ++k) acc = fmaf(x[k], W[k * dout + o], acc);
    if (use_softplus) acc = softplus_f(acc);
    Y[(size_t)r * dout + o] = acc;
  }
}

// ---------------------------------------------------------------------------
// Edge strain -> node stress accumulation.
// ---------------------------------------------------------------------------
__global__ __launch_bounds__(256)
void edge_stress_kernel(const float* __restrict__ disp, const float* __restrict__ pos,
                        const int* __restrict__ src, const int* __restrict__ dst,
                        float* __restrict__ ns, float* __restrict__ cnt, int E) {
  const int e = blockIdx.x * 256 + threadIdx.x;
  if (e >= E) return;
  const int s = src[e], d = dst[e];
  const float dux = disp[(size_t)d * 3 + 0] - disp[(size_t)s * 3 + 0];
  const float duy = disp[(size_t)d * 3 + 1] - disp[(size_t)s * 3 + 1];
  const float duz = disp[(size_t)d * 3 + 2] - disp[(size_t)s * 3 + 2];
  const float dxx = pos[(size_t)d * 3 + 0] - pos[(size_t)s * 3 + 0];
  const float dxy = pos[(size_t)d * 3 + 1] - pos[(size_t)s * 3 + 1];
  const float dxz = pos[(size_t)d * 3 + 2] - pos[(size_t)s * 3 + 2];
  const float dist = sqrtf(dxx * dxx + dxy * dxy + dxz * dxz) + 1e-8f;
  const float strain = sqrtf(dux * dux + duy * duy + duz * duz) / dist;
  const float es = 2.1e11f * strain;
  atomicAdd(ns + s, es);
  atomicAdd(ns + d, es);
  atomicAdd(cnt + s, 1.f);
  atomicAdd(cnt + d, 1.f);
}

__global__ __launch_bounds__(256)
void von_kernel(const float* __restrict__ ns, const float* __restrict__ cnt,
                float* __restrict__ out, int M) {
  const int r = blockIdx.x * 256 + threadIdx.x;
  if (r < M) out[r] = ns[r] / (cnt[r] + 1e-8f);
}

// ---------------------------------------------------------------------------
// Launch sequence.
// Input order: x, pos, edge_attr, edge_index, then params pytree leaves in
// key-sorted order (convs[0..2]{be,g,l1.b,l1.w,l2.b,l2.w}, disp_head{l1.b,
// l1.w,l2.b,l2.w}, edge_enc{...}, node_enc{...}, phys{...}, sf_head{...},
// stress_head{...}).
// Output: disp[N,3] | stress[N] | von_mises[N] | sf[N] | h[N,128].
// ---------------------------------------------------------------------------
extern "C" void kernel_launch(void* const* d_in, const int* in_sizes, int n_in,
                              void* d_out, int out_size, void* d_ws, size_t ws_size,
                              hipStream_t stream) {
  (void)in_sizes; (void)n_in; (void)out_size; (void)ws_size;

  const float* x     = (const float*)d_in[0];
  const float* pos   = (const float*)d_in[1];
  const float* eattr = (const float*)d_in[2];
  const int*   eidx  = (const int*)d_in[3];
  const int*   src   = eidx;
  const int*   dst   = eidx + NE;

  // Param leaf indices (see header comment).
  auto P = [&](int i) { return (const float*)d_in[i]; };
  const int CONV0 = 4;                 // +6 per conv: be,g,l1b,l1w,l2b,l2w
  const int DISP  = 22;                // l1b,l1w,l2b,l2w
  const int EENC  = 26;                // be,g,l1b,l1w,l2b,l2w
  const int NENC  = 32;
  const int PHYS  = 38;                // l1b,l1w,l2b,l2w
  const int SF    = 42;
  const int STRS  = 46;

  float* out   = (float*)d_out;
  float* disp  = out;                  // [N,3]
  float* strs  = out + (size_t)NN * 3; // [N]
  float* von   = strs + NN;            // [N]
  float* sf    = von + NN;             // [N]
  float* hout  = sf + NN;              // [N,128]

  // Workspace layout (fp32).
  float* ws   = (float*)d_ws;
  float* h    = ws;                          // N*128
  float* t1   = h    + (size_t)NN * 128;     // N*128
  float* t2   = t1   + (size_t)NN * 128;     // N*128
  float* aggr = t2   + (size_t)NN * 128;     // N*128
  float* pf   = aggr + (size_t)NN * 128;     // N*32
  float* ns   = pf   + (size_t)NN * 32;      // N
  float* cnt  = ns   + NN;                   // N
  float* eb1  = cnt  + NN;                   // E*128
  float* eb2  = eb1  + (size_t)NE * 128;     // E*128

  auto gemm_blocks = [](int M) {
    int tilesM = M >> 4;
    int blk = (tilesM + 7) >> 3;        // 8 waves per block, 1 row-block/wave
    return blk > 1024 ? 1024 : (blk < 1 ? 1 : blk);
  };

  const int nodeWaveBlk = (NN + 7) / 8;   // 1 wave/row kernels
  const int edgeWaveBlk = (NE + 7) / 8;
  const int nodeThrBlk  = (NN + 255) / 256;
  const int edgeThrBlk  = (NE + 255) / 256;

  // ---- node encoder: t1 = relu(x@W1+b1); t2 = t1@W2+b2; h = LN(t2)+pf ----
  small_linear_relu<<<(int)(((long long)NN * 128 + 255) / 256), 256, 0, stream>>>(
      x, P(NENC + 3), P(NENC + 2), t1, NN, 5);
  gemm_k128_wmma<<<gemm_blocks(NN), 256, 0, stream>>>(
      t1, P(NENC + 5), P(NENC + 4), t2, NN, 128, 0);
  phys_kernel<<<nodeWaveBlk, 256, 0, stream>>>(
      pos, P(PHYS + 1), P(PHYS + 0), P(PHYS + 3), P(PHYS + 2), pf, NN);
  ln_kernel<<<nodeWaveBlk, 256, 0, stream>>>(
      t2, h, P(NENC + 1), P(NENC + 0), pf, NN, 0);

  // ---- edge encoder: eb1 = relu(ea@W1+b1); eb2 = eb1@W2+b2; LN in-place ----
  small_linear_relu<<<(int)(((long long)NE * 128 + 255) / 256), 256, 0, stream>>>(
      eattr, P(EENC + 3), P(EENC + 2), eb1, NE, 6);
  gemm_k128_wmma<<<gemm_blocks(NE), 256, 0, stream>>>(
      eb1, P(EENC + 5), P(EENC + 4), eb2, NE, 128, 0);
  ln_kernel<<<edgeWaveBlk, 256, 0, stream>>>(
      eb2, eb2, P(EENC + 1), P(EENC + 0), nullptr, NE, 0);

  // ---- 3 GINE convs ----
  for (int ci = 0; ci < 3; ++ci) {
    const int CB = CONV0 + ci * 6;
    // aggr = h  (eps=0 => (1+eps)*x + aggr folds into init)
    hipMemcpyAsync(aggr, h, (size_t)NN * 128 * sizeof(float),
                   hipMemcpyDeviceToDevice, stream);
    message_kernel<<<edgeWaveBlk, 256, 0, stream>>>(h, eb2, src, dst, aggr, NE);
    gemm_k128_wmma<<<gemm_blocks(NN), 256, 0, stream>>>(
        aggr, P(CB + 3), P(CB + 2), t1, NN, 128, 1);          // l1 + relu
    gemm_k128_wmma<<<gemm_blocks(NN), 256, 0, stream>>>(
        t1, P(CB + 5), P(CB + 4), t2, NN, 128, 0);            // l2
    ln_kernel<<<nodeWaveBlk, 256, 0, stream>>>(
        t2, h, P(CB + 1), P(CB + 0), nullptr, NN, 1);          // LN + outer relu
  }

  // ---- heads (reuse t1 as [N,64] hidden) ----
  gemm_k128_wmma<<<gemm_blocks(NN), 256, 0, stream>>>(
      h, P(DISP + 1), P(DISP + 0), t1, NN, 64, 1);
  head_l2_kernel<<<nodeThrBlk, 256, 0, stream>>>(
      t1, P(DISP + 3), P(DISP + 2), disp, NN, 3, 0);

  gemm_k128_wmma<<<gemm_blocks(NN), 256, 0, stream>>>(
      h, P(STRS + 1), P(STRS + 0), t1, NN, 64, 1);
  head_l2_kernel<<<nodeThrBlk, 256, 0, stream>>>(
      t1, P(STRS + 3), P(STRS + 2), strs, NN, 1, 1);

  gemm_k128_wmma<<<gemm_blocks(NN), 256, 0, stream>>>(
      h, P(SF + 1), P(SF + 0), t1, NN, 64, 1);
  head_l2_kernel<<<nodeThrBlk, 256, 0, stream>>>(
      t1, P(SF + 3), P(SF + 2), sf, NN, 1, 1);

  // ---- von Mises from displacement gradients along edges ----
  hipMemsetAsync(ns, 0, (size_t)2 * NN * sizeof(float), stream);  // ns + cnt
  edge_stress_kernel<<<edgeThrBlk, 256, 0, stream>>>(
      disp, pos, src, dst, ns, cnt, NE);
  von_kernel<<<nodeThrBlk, 256, 0, stream>>>(ns, cnt, von, NN);

  // ---- final h to output ----
  hipMemcpyAsync(hout, h, (size_t)NN * 128 * sizeof(float),
                 hipMemcpyDeviceToDevice, stream);
}
